// GAT_47321949667761
// MI455X (gfx1250) — matrix operand order repaired and checked by reading, not compiled, not run on hardware
//
#include <hip/hip_runtime.h>
#include <hip/hip_bf16.h>
#include <math.h>

typedef __attribute__((ext_vector_type(16))) _Float16 v16h;
typedef __attribute__((ext_vector_type(8)))  _Float16 v8h;
typedef __attribute__((ext_vector_type(8)))  float    v8f;

#define LRELU_GAT 0.2f
#define LRELU_MLP 0.01f

// leaky relu with positive slope < 1:  lrelu(v) = max(v, slope*v)  (single v_max)
__device__ __forceinline__ float lrelu(float v, float slope) { return fmaxf(v, slope * v); }

// float atomic max via signed/unsigned integer atomics (init must be -inf)
__device__ __forceinline__ void atomicMaxF(float* addr, float val) {
  if (!(val < 0.0f)) {
    atomicMax((int*)addr, __float_as_int(val));
  } else {
    atomicMin((unsigned int*)addr, __float_as_uint(val));
  }
}

// ---------------------------------------------------------------------------
// GAT kernels (memory/atomic bound; VALU path). Templated on H,F,inF so the
// small feature loops fully unroll and gathers merge into wide loads.
// ---------------------------------------------------------------------------

__global__ void gat_init_kernel(float* __restrict__ smax, float* __restrict__ denom,
                                float* __restrict__ num, int nH, int nHF) {
  int i = blockIdx.x * blockDim.x + threadIdx.x;
  if (i < nH) { smax[i] = -INFINITY; denom[i] = 0.f; }
  if (i < nHF) num[i] = 0.f;
}

template <int H, int F, int INF>
__global__ void gat_node_feats_kernel(const float* __restrict__ in, int inStride,
                                      const float* __restrict__ Wsrc, const float* __restrict__ bsrc,
                                      const float* __restrict__ Wdst, const float* __restrict__ bdst,
                                      const float* __restrict__ Wres,
                                      float* __restrict__ fs, float* __restrict__ fd,
                                      float* __restrict__ res, int N) {
  constexpr int HF = H * F;
  int i = blockIdx.x * blockDim.x + threadIdx.x;
  if (i >= N) return;
  float xi[INF];
#pragma unroll
  for (int k = 0; k < INF; ++k) xi[k] = in[(size_t)i * inStride + k];
#pragma unroll
  for (int j = 0; j < HF; ++j) {
    float s = bsrc[j], d = bdst[j], r = 0.f;
#pragma unroll
    for (int k = 0; k < INF; ++k) {
      float v = xi[k];
      s += v * Wsrc[k * HF + j];
      d += v * Wdst[k * HF + j];
      r += v * Wres[k * HF + j];
    }
    fs[(size_t)i * HF + j] = s;
    fd[(size_t)i * HF + j] = d;
    res[(size_t)i * HF + j] = r;
  }
}

// pass 1: per-edge attention logit s, store it, segment-max into smax[dst]
template <int H, int F>
__global__ void gat_edge1_kernel(const int* __restrict__ src, const int* __restrict__ dst,
                                 const float* __restrict__ fs, const float* __restrict__ fd,
                                 const float* __restrict__ attn,
                                 float* __restrict__ sE, float* __restrict__ smax,
                                 long long E) {
  constexpr int HF = H * F;
  long long e = (long long)blockIdx.x * blockDim.x + threadIdx.x;
  if (e >= E) return;
  __builtin_prefetch(&src[e + 16384], 0, 0);   // streamed index arrays -> global_prefetch_b8
  __builtin_prefetch(&dst[e + 16384], 0, 0);
  int si = src[e], di = dst[e];
  float fsv[HF], fdv[HF];
#pragma unroll
  for (int j = 0; j < HF; ++j) fsv[j] = fs[(size_t)si * HF + j];
#pragma unroll
  for (int j = 0; j < HF; ++j) fdv[j] = fd[(size_t)di * HF + j];
#pragma unroll
  for (int h = 0; h < H; ++h) {
    float s = 0.f;
#pragma unroll
    for (int f = 0; f < F; ++f) {
      float v = lrelu(fsv[h * F + f] + fdv[h * F + f], LRELU_GAT);
      s += v * attn[h * F + f];
    }
    sE[(size_t)h * E + e] = s;
    atomicMaxF(&smax[(size_t)di * H + h], s);
  }
}

// pass 2: ex = exp(s - smax[dst]); denom[dst] += ex; num[dst] += ex * fs[src]
template <int H, int F>
__global__ void gat_edge2_kernel(const int* __restrict__ src, const int* __restrict__ dst,
                                 const float* __restrict__ fs,
                                 const float* __restrict__ sE, const float* __restrict__ smax,
                                 float* __restrict__ denom, float* __restrict__ num,
                                 long long E) {
  constexpr int HF = H * F;
  long long e = (long long)blockIdx.x * blockDim.x + threadIdx.x;
  if (e >= E) return;
  __builtin_prefetch(&src[e + 16384], 0, 0);
  __builtin_prefetch(&dst[e + 16384], 0, 0);
  int si = src[e], di = dst[e];
  float fsv[HF];
#pragma unroll
  for (int j = 0; j < HF; ++j) fsv[j] = fs[(size_t)si * HF + j];
#pragma unroll
  for (int h = 0; h < H; ++h) {
    float ex = __expf(sE[(size_t)h * E + e] - smax[(size_t)di * H + h]);
    atomicAdd(&denom[(size_t)di * H + h], ex);
#pragma unroll
    for (int f = 0; f < F; ++f) {
      atomicAdd(&num[(size_t)di * HF + h * F + f], ex * fsv[h * F + f]);
    }
  }
}

// out = elu(num/denom + res + bias)
template <int H, int F>
__global__ void gat_finalize_kernel(const float* __restrict__ num, const float* __restrict__ denom,
                                    const float* __restrict__ res, const float* __restrict__ bias,
                                    float* __restrict__ out, int N) {
  constexpr int HF = H * F;
  int i = blockIdx.x * blockDim.x + threadIdx.x;
  if (i >= N) return;
#pragma unroll
  for (int j = 0; j < HF; ++j) {
    float d = denom[(size_t)i * H + j / F];
    float r = (d > 0.f) ? num[(size_t)i * HF + j] / d : 0.f;
    r += res[(size_t)i * HF + j] + bias[j];
    out[(size_t)i * HF + j] = r > 0.f ? r : (__expf(r) - 1.f);
  }
}

// pack concat([h_att(2), h_def2(4), x(8)]) into zero-padded f16 [N,32]
__global__ void build_h14_kernel(const float* __restrict__ x, const float* __restrict__ hatt,
                                 const float* __restrict__ hdef2, _Float16* __restrict__ h14,
                                 int N) {
  int i = blockIdx.x * blockDim.x + threadIdx.x;
  if (i >= N) return;
  _Float16 row[32];
  int c = 0;
#pragma unroll
  for (int j = 0; j < 2; ++j) row[c++] = (_Float16)hatt[(size_t)i * 2 + j];
#pragma unroll
  for (int j = 0; j < 4; ++j) row[c++] = (_Float16)hdef2[(size_t)i * 4 + j];
#pragma unroll
  for (int j = 0; j < 8; ++j) row[c++] = (_Float16)x[(size_t)i * 8 + j];
#pragma unroll
  for (; c < 32; ++c) row[c] = (_Float16)0.f;
#pragma unroll
  for (int v = 0; v < 4; ++v)
    *(v8h*)(h14 + (size_t)i * 32 + v * 8) = *(const v8h*)(row + v * 8);
}

// ---------------------------------------------------------------------------
// MLP GEMM, transposed WMMA roles:
//   A = weight tile  (M = 16 out-features x 32 K), pre-swizzled in LDS
//   B = activations  (32 K x 16 nodes), one contiguous 32B global load/chunk
//   D: lane = node, vgpr r = out-feature  ->  epilogue is ONE 16B b128 store
// Activation buffers are padded to Nr rows, so no row guards anywhere.
// ---------------------------------------------------------------------------
template <int KPAD, int MT_BLK>
__global__ __launch_bounds__(256) void mlp_gemm_kernel(const _Float16* __restrict__ in, int K,
                                                       const float* __restrict__ W, int M,
                                                       const float* __restrict__ bias,
                                                       _Float16* __restrict__ out, int outStride) {
  constexpr int NKC = KPAD / 32;                        // K chunks of 32
  __shared__ __align__(32) _Float16 wlds[MT_BLK * NKC * 512];
  __shared__ __align__(32) float blds[MT_BLK * 16];

  const int tid = threadIdx.x;
  // Weights -> A-fragment layout. contiguous K order used here (B matches):
  //   lane<16 : e0..15 -> K 0..15   (row m = lane&15)
  //   lane>=16: e0..15 -> K 16..31
  for (int idx = tid; idx < MT_BLK * NKC * 512; idx += 256) {
    int e    = idx & 15;
    int lane = (idx >> 4) & 31;
    int kc   = (idx >> 9) % NKC;
    int mtl  = idx / (512 * NKC);
    int m = (blockIdx.y * MT_BLK + mtl) * 16 + (lane & 15);
    int k = kc * 32 + ((lane & 16) ? 16 : 0) + e;
    wlds[idx] = (k < K && m < M) ? (_Float16)W[(size_t)k * M + m] : (_Float16)0.f;
  }
  // Bias, zero-padded, so epilogue has no bounds branch.
  for (int j = tid; j < MT_BLK * 16; j += 256) {
    int m = blockIdx.y * MT_BLK * 16 + j;
    blds[j] = (m < M) ? bias[m] : 0.f;
  }
  __syncthreads();

  const int lane = tid & 31;
  const int wave = tid >> 5;
  const int row_base = (blockIdx.x * 8 + wave) * 16;    // 16 node rows per wave
  const int node = row_base + (lane & 15);

  // B fragments: lane holds node column (lane&15); elements = contiguous K run
  // (lane<16 -> K kb..kb+15, lane>=16 -> K kb+16..kb+31): one 32B load/chunk.
  const _Float16* inRow = in + (size_t)node * KPAD + ((lane & 16) ? 16 : 0);
  v16h bfrag[NKC];
#pragma unroll
  for (int kc = 0; kc < NKC; ++kc) {
    bfrag[kc] = *(const v16h*)(inRow + kc * 32);
  }

#pragma unroll
  for (int mtl = 0; mtl < MT_BLK; ++mtl) {
    v8f acc = {0.f, 0.f, 0.f, 0.f, 0.f, 0.f, 0.f, 0.f};
#pragma unroll
    for (int kc = 0; kc < NKC; ++kc) {
      v16h afrag = *(const v16h*)(&wlds[((mtl * NKC + kc) * 32 + lane) * 16]);
      acc = __builtin_amdgcn_wmma_f32_16x16x32_f16(false, afrag, false, bfrag[kc],
                                                   (short)0, acc, false, false);
    }
    // D: lane = node col; vgpr r = out-feature r + (lane<16 ? 0 : 8)
    const int fo = (lane & 16) ? 8 : 0;
    const int obase = (blockIdx.y * MT_BLK + mtl) * 16 + fo;
    const float* bp = &blds[mtl * 16 + fo];
    v8h ov;
#pragma unroll
    for (int r = 0; r < 8; ++r) {
      float v = lrelu(acc[r] + bp[r], LRELU_MLP);   // v_max_num_f32, branchless
      ov[r] = (_Float16)v;
    }
    *(v8h*)(out + (size_t)node * outStride + obase) = ov;   // single 16B store
  }
}

// final 14 -> 1 dot + sigmoid, f32 output
__global__ void mlp_final_kernel(const _Float16* __restrict__ h3, const float* __restrict__ W4,
                                 const float* __restrict__ b4, float* __restrict__ out, int N) {
  int i = blockIdx.x * blockDim.x + threadIdx.x;
  if (i >= N) return;
  float s = b4[0];
#pragma unroll
  for (int k = 0; k < 14; ++k) s += (float)h3[(size_t)i * 32 + k] * W4[k];
  out[i] = 1.f / (1.f + __expf(-s));
}

// ---------------------------------------------------------------------------
// Host driver
// ---------------------------------------------------------------------------

template <int H, int F, int INF>
static void run_gat(const float* in, int inStride,
                    const int* src, const int* dst, long long E, int N,
                    const float* Wsrc, const float* bsrc, const float* Wdst, const float* bdst,
                    const float* attn, const float* Wres, const float* bias,
                    float* fs, float* fd, float* res, float* smax, float* denom, float* num,
                    float* sE, float* out, hipStream_t stream) {
  constexpr int HF = H * F;
  gat_init_kernel<<<dim3((unsigned)(((size_t)N * HF + 255) / 256)), 256, 0, stream>>>(
      smax, denom, num, N * H, N * HF);
  gat_node_feats_kernel<H, F, INF><<<dim3((unsigned)((N + 255) / 256)), 256, 0, stream>>>(
      in, inStride, Wsrc, bsrc, Wdst, bdst, Wres, fs, fd, res, N);
  unsigned eBlocks = (unsigned)((E + 255) / 256);
  gat_edge1_kernel<H, F><<<dim3(eBlocks), 256, 0, stream>>>(src, dst, fs, fd, attn, sE, smax, E);
  gat_edge2_kernel<H, F><<<dim3(eBlocks), 256, 0, stream>>>(src, dst, fs, sE, smax, denom, num, E);
  gat_finalize_kernel<H, F><<<dim3((unsigned)((N + 255) / 256)), 256, 0, stream>>>(
      num, denom, res, bias, out, N);
}

extern "C" void kernel_launch(void* const* d_in, const int* in_sizes, int n_in,
                              void* d_out, int out_size, void* d_ws, size_t ws_size,
                              hipStream_t stream) {
  const float* x   = (const float*)d_in[0];
  const int*   src = (const int*)d_in[1];
  const int*   dst = (const int*)d_in[2];

  const float* a1_Wsrc = (const float*)d_in[3];
  const float* a1_bsrc = (const float*)d_in[4];
  const float* a1_Wdst = (const float*)d_in[5];
  const float* a1_bdst = (const float*)d_in[6];
  const float* a1_attn = (const float*)d_in[7];
  const float* a1_Wres = (const float*)d_in[8];
  const float* a1_bias = (const float*)d_in[9];

  const float* d1_Wsrc = (const float*)d_in[10];
  const float* d1_bsrc = (const float*)d_in[11];
  const float* d1_Wdst = (const float*)d_in[12];
  const float* d1_bdst = (const float*)d_in[13];
  const float* d1_attn = (const float*)d_in[14];
  const float* d1_Wres = (const float*)d_in[15];
  const float* d1_bias = (const float*)d_in[16];

  const float* d2_Wsrc = (const float*)d_in[17];
  const float* d2_bsrc = (const float*)d_in[18];
  const float* d2_Wdst = (const float*)d_in[19];
  const float* d2_bdst = (const float*)d_in[20];
  const float* d2_attn = (const float*)d_in[21];
  const float* d2_Wres = (const float*)d_in[22];
  const float* d2_bias = (const float*)d_in[23];

  const float* W1 = (const float*)d_in[24];
  const float* b1 = (const float*)d_in[25];
  const float* W2 = (const float*)d_in[26];
  const float* b2 = (const float*)d_in[27];
  const float* W3 = (const float*)d_in[28];
  const float* b3 = (const float*)d_in[29];
  const float* W4 = (const float*)d_in[30];
  const float* b4 = (const float*)d_in[31];

  const int N = in_sizes[0] / 8;
  const long long E = in_sizes[1];
  const int Nr = ((N + 127) / 128) * 128;   // rows padded to 8 waves * 16

  // bump allocator over workspace, 256B aligned
  char* ws = (char*)d_ws;
  size_t off = 0;
  auto alloc = [&](size_t bytes) -> void* {
    void* p = ws + off;
    off = (off + bytes + 255) & ~(size_t)255;
    return p;
  };

  float* fs    = (float*)alloc(sizeof(float) * (size_t)Nr * 10);
  float* fd    = (float*)alloc(sizeof(float) * (size_t)Nr * 10);
  float* res   = (float*)alloc(sizeof(float) * (size_t)Nr * 10);
  float* smax  = (float*)alloc(sizeof(float) * (size_t)Nr * 2);
  float* denom = (float*)alloc(sizeof(float) * (size_t)Nr * 2);
  float* num   = (float*)alloc(sizeof(float) * (size_t)Nr * 10);
  float* sE    = (float*)alloc(sizeof(float) * (size_t)E * 2);
  float* h_att = (float*)alloc(sizeof(float) * (size_t)Nr * 2);
  float* hdef1 = (float*)alloc(sizeof(float) * (size_t)Nr * 10);
  float* hdef2 = (float*)alloc(sizeof(float) * (size_t)Nr * 4);
  _Float16* h14p = (_Float16*)alloc(sizeof(_Float16) * (size_t)Nr * 32);
  _Float16* h1p  = (_Float16*)alloc(sizeof(_Float16) * (size_t)Nr * 224);
  _Float16* h2p  = (_Float16*)alloc(sizeof(_Float16) * (size_t)Nr * 224);
  _Float16* h3p  = (_Float16*)alloc(sizeof(_Float16) * (size_t)Nr * 32);
  (void)ws_size; (void)n_in; (void)out_size;

  // --- GAT layers (g_in = x[:, :5], stride 8) ---
  run_gat<2, 1, 5>(x, 8, src, dst, E, N, a1_Wsrc, a1_bsrc, a1_Wdst, a1_bdst, a1_attn, a1_Wres,
                   a1_bias, fs, fd, res, smax, denom, num, sE, h_att, stream);
  run_gat<2, 5, 5>(x, 8, src, dst, E, N, d1_Wsrc, d1_bsrc, d1_Wdst, d1_bdst, d1_attn, d1_Wres,
                   d1_bias, fs, fd, res, smax, denom, num, sE, hdef1, stream);
  run_gat<2, 2, 10>(hdef1, 10, src, dst, E, N, d2_Wsrc, d2_bsrc, d2_Wdst, d2_bdst, d2_attn,
                    d2_Wres, d2_bias, fs, fd, res, smax, denom, num, sE, hdef2, stream);

  // --- pack MLP input ---
  build_h14_kernel<<<dim3((unsigned)((N + 255) / 256)), 256, 0, stream>>>(x, h_att, hdef2, h14p, N);

  // --- WMMA MLP (weights-as-A, activations-as-B) ---
  const unsigned rowBlocks = (unsigned)(Nr / 128);
  // layer 1: K=14 (pad 32), M=196 (pad 224): 2 y-blocks * 7 m-tiles
  mlp_gemm_kernel<32, 7><<<dim3(rowBlocks, 2), 256, 0, stream>>>(h14p, 14, W1, 196, b1, h1p, 224);
  // layer 2: K=196 (pad 224), M=196 (pad 224)
  mlp_gemm_kernel<224, 7><<<dim3(rowBlocks, 2), 256, 0, stream>>>(h1p, 196, W2, 196, b2, h2p, 224);
  // layer 3: K=196 (pad 224), M=14 (pad 32): 1 y-block * 2 m-tiles
  mlp_gemm_kernel<224, 2><<<dim3(rowBlocks, 1), 256, 0, stream>>>(h2p, 196, W3, 14, b3, h3p, 32);
  // layer 4 + sigmoid
  mlp_final_kernel<<<dim3((unsigned)((N + 255) / 256)), 256, 0, stream>>>(h3p, W4, b4,
                                                                          (float*)d_out, N);
}
